// QuantizedAttention_39900246180185
// MI455X (gfx1250) — compile-verified, matrix operand order checked by reading, not run on hardware
//
#include <hip/hip_runtime.h>
#include <stdint.h>

// ---------------- problem constants ----------------
constexpr int B_ = 2, S_ = 1024, P_ = 1024, D_ = 2048, H_ = 16;
constexpr int T_ = P_ + S_;           // 2048
constexpr int DH_ = D_ / H_;          // 128
constexpr int BSD = B_ * S_ * D_;     // 4194304
constexpr int BPD = B_ * P_ * D_;     // 4194304
constexpr float INV_SQRT_DH = 0.0883883476483184f; // 1/sqrt(128)

typedef __bf16 v16bf __attribute__((ext_vector_type(16)));
typedef float  v8f   __attribute__((ext_vector_type(8)));

__device__ __forceinline__ unsigned short f2bf(float f) {
  unsigned int u = __float_as_uint(f);
  u += 0x7FFFu + ((u >> 16) & 1u);      // round-to-nearest-even
  return (unsigned short)(u >> 16);
}
__device__ __forceinline__ __bf16 us2bf(unsigned short u) {
  return __builtin_bit_cast(__bf16, u);
}
// K-pair base for 16-bit A/B fragment VGPR i, K-half h (CDNA5 WMMA layout)
__device__ __forceinline__ int kbase(int i, int half) {
  return (i < 4 ? 2 * i : 16 + 2 * (i - 4)) + 8 * half;
}

// ---------------- elementwise converters ----------------
__global__ void f32_to_bf16_kernel(const float* __restrict__ in,
                                   unsigned short* __restrict__ out, int n) {
  int i = blockIdx.x * blockDim.x + threadIdx.x;
  if (i < n) out[i] = f2bf(in[i]);
}

// out[n][k] = bf16(in[k][n])  (D_ x D_ square)
__global__ void f32_to_bf16_T_kernel(const float* __restrict__ in,
                                     unsigned short* __restrict__ out, int n) {
  int i = blockIdx.x * blockDim.x + threadIdx.x;
  if (i >= n) return;
  int nc = i / D_;
  int k  = i % D_;
  out[i] = f2bf(in[(size_t)k * D_ + nc]);
}

// ---------------- generic bf16 WMMA GEMM ----------------
// C[M,N] = A[M,K](bf16) * Bt[N,K](bf16, B transposed) + bias[N]
// one wave computes a 32x64 tile (2 M-subtiles x 4 N-subtiles); K-step 32.
// B fragments are reused across both M-subtiles -> 6 fragment loads / 8 WMMAs.
// Prefetches are speculative and the
// one-step-past-the-end addresses stay inside the workspace, so no guards.
__global__ __launch_bounds__(128)
void gemm_bf16_wmma(const unsigned short* __restrict__ A,
                    const unsigned short* __restrict__ Bt,
                    const float* __restrict__ bias,
                    float* __restrict__ Cf,
                    unsigned short* __restrict__ Cb,
                    int M, int N, int K, float outScale) {
  int wid  = threadIdx.x >> 5;
  int lane = threadIdx.x & 31;
  int half = lane >> 4, ln = lane & 15;
  int w = blockIdx.x * 4 + wid;
  int ntiles = N / 64;
  int wm = (w / ntiles) * 32;
  int wn = (w % ntiles) * 64;
  if (wm >= M) return;

  v8f acc[2][4] = {};
  const unsigned short* arow[2] = {A + (size_t)(wm + ln) * K,
                                   A + (size_t)(wm + 16 + ln) * K};

  for (int k0 = 0; k0 < K; k0 += 32) {
    v16bf af[2];
#pragma unroll
    for (int mi = 0; mi < 2; ++mi) {
#pragma unroll
      for (int i = 0; i < 8; ++i) {
        int kb = kbase(i, half);
        unsigned int pv = *(const unsigned int*)(arow[mi] + k0 + kb);
        af[mi][2 * i]     = us2bf((unsigned short)(pv & 0xffffu));
        af[mi][2 * i + 1] = us2bf((unsigned short)(pv >> 16));
      }
      __builtin_prefetch(arow[mi] + k0 + 32, 0, 3);
    }
#pragma unroll
    for (int j = 0; j < 4; ++j) {
      const unsigned short* brow = Bt + (size_t)(wn + j * 16 + ln) * K + k0;
      v16bf bf_;
#pragma unroll
      for (int i = 0; i < 8; ++i) {
        int kb = kbase(i, half);
        unsigned int pv = *(const unsigned int*)(brow + kb);
        bf_[2 * i]     = us2bf((unsigned short)(pv & 0xffffu));
        bf_[2 * i + 1] = us2bf((unsigned short)(pv >> 16));
      }
      __builtin_prefetch(brow + 32, 0, 3);
      acc[0][j] = __builtin_amdgcn_wmma_f32_16x16x32_bf16(
          false, af[0], false, bf_, (short)0, acc[0][j], false, false);
      acc[1][j] = __builtin_amdgcn_wmma_f32_16x16x32_bf16(
          false, af[1], false, bf_, (short)0, acc[1][j], false, false);
    }
  }

#pragma unroll
  for (int mi = 0; mi < 2; ++mi)
#pragma unroll
    for (int j = 0; j < 4; ++j)
#pragma unroll
      for (int i = 0; i < 8; ++i) {
        int row = wm + mi * 16 + i + 8 * half;
        int col = wn + j * 16 + ln;
        float o = acc[mi][j][i] + bias[col];
        if (Cb) Cb[(size_t)row * N + col] = f2bf(o * outScale);
        else    Cf[(size_t)row * N + col] = o;
      }
}

// ---------------- abs-max reduction ----------------
__global__ void init_scalars_kernel(unsigned int* p) {
  p[0] = 0u; p[1] = 0u;
}

__global__ void absmax_kernel(const float* __restrict__ x, int n,
                              unsigned int* __restrict__ result) {
  __shared__ float red[256];
  float m = 0.f;
  for (int i = blockIdx.x * blockDim.x + threadIdx.x; i < n;
       i += gridDim.x * blockDim.x)
    m = fmaxf(m, fabsf(x[i]));
  red[threadIdx.x] = m;
  __syncthreads();
  for (int s = 128; s > 0; s >>= 1) {
    if (threadIdx.x < s)
      red[threadIdx.x] = fmaxf(red[threadIdx.x], red[threadIdx.x + s]);
    __syncthreads();
  }
  if (threadIdx.x == 0) atomicMax(result, __float_as_uint(red[0]));
}

// ---------------- quantize current k/v + stage raw bf16 into cache ----------------
// mode 0: full is kfull [B,T,D] row-major; mode 1: full is vfullT [B,D,T]
__global__ void quantize_kernel(const float* __restrict__ x,
                                const unsigned int* __restrict__ amax_bits,
                                float* __restrict__ qout,
                                float* __restrict__ scale_out,
                                unsigned short* __restrict__ full, int mode) {
  int i = blockIdx.x * blockDim.x + threadIdx.x;
  float amax  = __uint_as_float(*amax_bits);
  float scale = amax / 127.0f;
  if (i == 0) *scale_out = scale;
  if (i >= BSD) return;
  float v = x[i];
  float q = rintf(v / scale);
  q = fminf(fmaxf(q, -128.f), 127.f);
  qout[i] = q;
  int d = i % D_;
  int s = (i / D_) % S_;
  int b = i / (D_ * S_);
  if (mode == 0) full[((size_t)b * T_ + P_ + s) * D_ + d] = f2bf(v);
  else           full[((size_t)b * D_ + d) * T_ + P_ + s] = f2bf(v);
}

// dequantize int8 KV cache into the concatenated bf16 buffers
__global__ void dequant_past_kernel(const int* __restrict__ pq,
                                    const float* __restrict__ scale,
                                    unsigned short* __restrict__ full, int mode) {
  int i = blockIdx.x * blockDim.x + threadIdx.x;
  if (i >= BPD) return;
  float v = (float)pq[i] * (*scale);
  int d = i % D_;
  int t = (i / D_) % P_;
  int b = i / (D_ * P_);
  if (mode == 0) full[((size_t)b * T_ + t) * D_ + d] = f2bf(v);
  else           full[((size_t)b * D_ + d) * T_ + t] = f2bf(v);
}

// ---------------- flash attention, 1 wave per (b,h,16-row q block) ----------------
__global__ __launch_bounds__(128)
void attn_wmma_kernel(const unsigned short* __restrict__ qbf,    // [B,S,D], pre-scaled
                      const unsigned short* __restrict__ kfull,  // [B,T,D]
                      const unsigned short* __restrict__ vfullT, // [B,D,T]
                      unsigned short* __restrict__ obf) {        // [B,S,D]
  __shared__ alignas(16) unsigned short pst[4][16 * 32];
  int wid  = threadIdx.x >> 5;
  int lane = threadIdx.x & 31;
  int half = lane >> 4, ln = lane & 15;
  int gw  = blockIdx.x * 4 + wid;            // 0..2047
  int q16 = gw % (S_ / 16);
  int bh  = gw / (S_ / 16);
  int hh  = bh % H_;
  int b   = bh / H_;
  int s0  = q16 * 16;

  // Q fragments: 4 K-chunks of 32 covering DH=128
  v16bf qf[4];
  const unsigned short* qrow = qbf + ((size_t)b * S_ + (s0 + ln)) * D_ + hh * DH_;
#pragma unroll
  for (int c = 0; c < 4; ++c)
#pragma unroll
    for (int i = 0; i < 8; ++i) {
      int kb = kbase(i, half);
      unsigned int pv = *(const unsigned int*)(qrow + c * 32 + kb);
      qf[c][2 * i]     = us2bf((unsigned short)(pv & 0xffffu));
      qf[c][2 * i + 1] = us2bf((unsigned short)(pv >> 16));
    }

  v8f acc[8] = {};
  float mrow[8], lrow[8];
#pragma unroll
  for (int i = 0; i < 8; ++i) { mrow[i] = -1e30f; lrow[i] = 0.f; }

  int tlimit = P_ + s0 + 16;
  if (tlimit > T_) tlimit = T_;

  for (int t0 = 0; t0 < tlimit; t0 += 32) {
    // scores: two 16x16 tiles over this 32-column chunk
    v8f sc[2];
#pragma unroll
    for (int sub = 0; sub < 2; ++sub) {
      sc[sub] = (v8f){};
      int tcol = t0 + sub * 16 + ln;
      const unsigned short* krow = kfull + ((size_t)b * T_ + tcol) * D_ + hh * DH_;
      __builtin_prefetch(krow + (size_t)32 * D_, 0, 3); // next chunk's K row
#pragma unroll
      for (int c = 0; c < 4; ++c) {
        v16bf kf;
#pragma unroll
        for (int i = 0; i < 8; ++i) {
          int kb = kbase(i, half);
          unsigned int pv = *(const unsigned int*)(krow + c * 32 + kb);
          kf[2 * i]     = us2bf((unsigned short)(pv & 0xffffu));
          kf[2 * i + 1] = us2bf((unsigned short)(pv >> 16));
        }
        sc[sub] = __builtin_amdgcn_wmma_f32_16x16x32_bf16(
            false, qf[c], false, kf, (short)0, sc[sub], false, false);
      }
    }

    // online softmax per row (rows of C-tile: M = i + 8*half, N = lane&15)
#pragma unroll
    for (int i = 0; i < 8; ++i) {
      int row  = i + 8 * half;
      int slim = P_ + s0 + row;           // causal: t <= slim allowed
      float v0 = sc[0][i]; if (t0 + ln > slim)      v0 = -1e30f;
      float v1 = sc[1][i]; if (t0 + 16 + ln > slim) v1 = -1e30f;
      float mx = fmaxf(v0, v1);
      mx = fmaxf(mx, __shfl_xor(mx, 1));
      mx = fmaxf(mx, __shfl_xor(mx, 2));
      mx = fmaxf(mx, __shfl_xor(mx, 4));
      mx = fmaxf(mx, __shfl_xor(mx, 8));
      float mnew = fmaxf(mrow[i], mx);
      float p0 = __expf(v0 - mnew);
      float p1 = __expf(v1 - mnew);
      float rs = p0 + p1;
      rs += __shfl_xor(rs, 1);
      rs += __shfl_xor(rs, 2);
      rs += __shfl_xor(rs, 4);
      rs += __shfl_xor(rs, 8);
      float corr = __expf(mrow[i] - mnew);
      lrow[i] = lrow[i] * corr + rs;
      mrow[i] = mnew;
#pragma unroll
      for (int d = 0; d < 8; ++d) acc[d][i] *= corr;
      pst[wid][row * 32 + ln]      = f2bf(p0);
      pst[wid][row * 32 + 16 + ln] = f2bf(p1);
    }

    asm volatile("s_wait_dscnt 0" ::: "memory");

    // re-layout probabilities C-tile -> A-fragment (16x32)
    v16bf pf;
#pragma unroll
    for (int i = 0; i < 8; ++i) {
      int kb = kbase(i, half);
      unsigned int pv = *(const unsigned int*)&pst[wid][ln * 32 + kb];
      pf[2 * i]     = us2bf((unsigned short)(pv & 0xffffu));
      pf[2 * i + 1] = us2bf((unsigned short)(pv >> 16));
    }

    // P @ V : 8 dh-tiles of 16 columns; V stored transposed -> contiguous pairs
#pragma unroll
    for (int d = 0; d < 8; ++d) {
      const unsigned short* vcol =
          vfullT + ((size_t)b * D_ + hh * DH_ + d * 16 + ln) * T_ + t0;
      __builtin_prefetch(vcol + 32, 0, 3); // next chunk's V columns
      v16bf vf;
#pragma unroll
      for (int i = 0; i < 8; ++i) {
        int kb = kbase(i, half);
        unsigned int pv = *(const unsigned int*)(vcol + kb);
        vf[2 * i]     = us2bf((unsigned short)(pv & 0xffffu));
        vf[2 * i + 1] = us2bf((unsigned short)(pv >> 16));
      }
      acc[d] = __builtin_amdgcn_wmma_f32_16x16x32_bf16(
          false, pf, false, vf, (short)0, acc[d], false, false);
    }
  }

  // normalize and store bf16 attention output
#pragma unroll
  for (int d = 0; d < 8; ++d)
#pragma unroll
    for (int i = 0; i < 8; ++i) {
      int row = i + 8 * half;
      float o = acc[d][i] / lrow[i];
      obf[((size_t)b * S_ + s0 + row) * D_ + hh * DH_ + d * 16 + ln] = f2bf(o);
    }
}

// ---------------- host orchestration ----------------
extern "C" void kernel_launch(void* const* d_in, const int* in_sizes, int n_in,
                              void* d_out, int out_size, void* d_ws, size_t ws_size,
                              hipStream_t stream) {
  (void)in_sizes; (void)n_in; (void)out_size; (void)ws_size;

  const float* h   = (const float*)d_in[0];
  const float* Wq  = (const float*)d_in[1];
  const float* bq  = (const float*)d_in[2];
  const float* Wk  = (const float*)d_in[3];
  const float* bk  = (const float*)d_in[4];
  const float* Wv  = (const float*)d_in[5];
  const float* bv  = (const float*)d_in[6];
  const float* Wo  = (const float*)d_in[7];
  const float* bo  = (const float*)d_in[8];
  const int*   pkq = (const int*)d_in[9];
  const float* pks = (const float*)d_in[10];
  const int*   pvq = (const int*)d_in[11];
  const float* pvs = (const float*)d_in[12];

  // output regions: out | k_q | k_scale | v_q | v_scale (all as float)
  float* out_attn   = (float*)d_out;
  float* out_kq     = out_attn + BSD;
  float* out_kscale = out_kq + BSD;
  float* out_vq     = out_kscale + 1;
  float* out_vscale = out_vq + BSD;

  // workspace carve-up (~120 MB)
  char* w = (char*)d_ws;
  auto alloc = [&](size_t bytes) {
    char* p = w;
    w += (bytes + 255) & ~(size_t)255;
    return p;
  };
  unsigned short* hbf    = (unsigned short*)alloc((size_t)BSD * 2);
  unsigned short* WqT    = (unsigned short*)alloc((size_t)D_ * D_ * 2);
  unsigned short* WkT    = (unsigned short*)alloc((size_t)D_ * D_ * 2);
  unsigned short* WvT    = (unsigned short*)alloc((size_t)D_ * D_ * 2);
  unsigned short* WoT    = (unsigned short*)alloc((size_t)D_ * D_ * 2);
  unsigned short* qbf    = (unsigned short*)alloc((size_t)BSD * 2);
  float*          kf     = (float*)alloc((size_t)BSD * 4);
  float*          vf     = (float*)alloc((size_t)BSD * 4);
  unsigned short* kfull  = (unsigned short*)alloc((size_t)B_ * T_ * D_ * 2);
  unsigned short* vfullT = (unsigned short*)alloc((size_t)B_ * T_ * D_ * 2);
  unsigned short* attnbf = (unsigned short*)alloc((size_t)BSD * 2);
  unsigned int*   maxbits = (unsigned int*)alloc(256); // [0]=kmax, [1]=vmax

  const int EW_BLK = 256;
  dim3 ewGrid((BSD + EW_BLK - 1) / EW_BLK);
  const int M = B_ * S_, N = D_, K = D_;
  dim3 gemmGrid((M / 32) * (N / 64) / 4);   // 512 blocks x 4 waves (32x64 tiles)
  dim3 attnGrid(B_ * H_ * (S_ / 16) / 4);   // 512 blocks x 4 waves

  // 1. bf16 conversions (h plain, weights transposed for contiguous B-fragments)
  f32_to_bf16_kernel<<<ewGrid, EW_BLK, 0, stream>>>(h, hbf, BSD);
  f32_to_bf16_T_kernel<<<ewGrid, EW_BLK, 0, stream>>>(Wq, WqT, D_ * D_);
  f32_to_bf16_T_kernel<<<ewGrid, EW_BLK, 0, stream>>>(Wk, WkT, D_ * D_);
  f32_to_bf16_T_kernel<<<ewGrid, EW_BLK, 0, stream>>>(Wv, WvT, D_ * D_);
  f32_to_bf16_T_kernel<<<ewGrid, EW_BLK, 0, stream>>>(Wo, WoT, D_ * D_);

  // 2. projections (K,V fp32 for quantization; Q straight to scaled bf16)
  gemm_bf16_wmma<<<gemmGrid, 128, 0, stream>>>(hbf, WkT, bk, kf, nullptr, M, N, K, 1.f);
  gemm_bf16_wmma<<<gemmGrid, 128, 0, stream>>>(hbf, WvT, bv, vf, nullptr, M, N, K, 1.f);
  gemm_bf16_wmma<<<gemmGrid, 128, 0, stream>>>(hbf, WqT, bq, nullptr, qbf, M, N, K, INV_SQRT_DH);

  // 3. per-tensor abs-max
  init_scalars_kernel<<<1, 1, 0, stream>>>(maxbits);
  absmax_kernel<<<1024, 256, 0, stream>>>(kf, BSD, maxbits + 0);
  absmax_kernel<<<1024, 256, 0, stream>>>(vf, BSD, maxbits + 1);

  // 4. quantize current k/v (outputs) + stage raw bf16 into concat caches
  quantize_kernel<<<ewGrid, EW_BLK, 0, stream>>>(kf, maxbits + 0, out_kq, out_kscale, kfull, 0);
  quantize_kernel<<<ewGrid, EW_BLK, 0, stream>>>(vf, maxbits + 1, out_vq, out_vscale, vfullT, 1);

  // 5. dequantize int8 KV cache into caches (v transposed)
  dequant_past_kernel<<<ewGrid, EW_BLK, 0, stream>>>(pkq, pks, kfull, 0);
  dequant_past_kernel<<<ewGrid, EW_BLK, 0, stream>>>(pvq, pvs, vfullT, 1);

  // 6. flash attention (bf16 WMMA for q@k^T and p@v)
  attn_wmma_kernel<<<attnGrid, 128, 0, stream>>>(qbf, kfull, vfullT, attnbf);

  // 7. output projection
  gemm_bf16_wmma<<<gemmGrid, 128, 0, stream>>>(attnbf, WoT, bo, out_attn, nullptr, M, N, K, 1.f);
}
